// GatConvEmbedding_33741263077929
// MI455X (gfx1250) — matrix operand order repaired and checked by reading, not compiled, not run on hardware
//
#include <hip/hip_runtime.h>
#include <hip/hip_bf16.h>
#include <math.h>

typedef _Float16 v16h __attribute__((ext_vector_type(16)));
typedef float    v8f  __attribute__((ext_vector_type(8)));

#define DOUT 64
#define TPB 256

// ---------------------------------------------------------------------------
// float atomic max via monotonic int trick (works for mixed signs)
// ---------------------------------------------------------------------------
__device__ __forceinline__ void atomicMaxFloat(float* addr, float val) {
    if (val >= 0.0f) {
        atomicMax((int*)addr, __float_as_int(val));
    } else {
        atomicMin((unsigned int*)addr, __float_as_uint(val));
    }
}

// ---------------------------------------------------------------------------
// GEMM + bias:  out[M x 64] = A[M x K] * W[K x 64] + bias
// One wave computes a 16 x 64 stripe: 4 WMMA accumulators (16x16 each),
// K-loop in steps of 32 via v_wmma_f32_16x16x32_f16.
// A-fragment layout (16-bit A 16x32): lane<16 -> K 0..7 & 16..23,
// lane>=16 -> K 8..15 & 24..31, row M = lane&15.
// B-fragment layout (16-bit B 32x16): column N = lane&15, K same as A fragment.
// ---------------------------------------------------------------------------
__global__ void gemm_bias_wmma(const float* __restrict__ A,
                               const float* __restrict__ W,
                               const float* __restrict__ bias,
                               float* __restrict__ out,
                               int M, int K) {
    int wave = (int)((blockIdx.x * blockDim.x + threadIdx.x) >> 5);
    int lane = (int)(threadIdx.x & 31);
    int m0 = wave * 16;
    if (m0 >= M) return;                       // wave-uniform: EXEC stays full

    int rowA = m0 + (lane & 15);
    if (rowA >= M) rowA = M - 1;               // clamp reads (M%16==0 normally)
    int koff = (lane & 16) ? 8 : 0;

    v8f c0 = {}, c1 = {}, c2 = {}, c3 = {};

    for (int k0 = 0; k0 < K; k0 += 32) {
        // ---- load A fragment (f32 -> f16) ----
        const float* arow = A + (size_t)rowA * K + k0 + koff;
        v16h a;
#pragma unroll
        for (int i = 0; i < 8; ++i) {
            a[i]     = (_Float16)arow[i];        // K = kbase + 0..7
            a[8 + i] = (_Float16)arow[16 + i];   // K = kbase + 16..23
        }
        // ---- B fragments: lane holds column N = lane&15, K as in A fragment ----
        const float* wrow = W + (size_t)(k0 + koff) * DOUT + (lane & 15);
        v16h b0, b1, b2, b3;
#pragma unroll
        for (int n = 0; n < 8; ++n) {
            b0[n] = (_Float16)wrow[n * DOUT];
            b1[n] = (_Float16)wrow[n * DOUT + 16];
            b2[n] = (_Float16)wrow[n * DOUT + 32];
            b3[n] = (_Float16)wrow[n * DOUT + 48];
            b0[8 + n] = (_Float16)wrow[(16 + n) * DOUT];
            b1[8 + n] = (_Float16)wrow[(16 + n) * DOUT + 16];
            b2[8 + n] = (_Float16)wrow[(16 + n) * DOUT + 32];
            b3[8 + n] = (_Float16)wrow[(16 + n) * DOUT + 48];
        }
        c0 = __builtin_amdgcn_wmma_f32_16x16x32_f16(false, a, false, b0, (short)0, c0, false, false);
        c1 = __builtin_amdgcn_wmma_f32_16x16x32_f16(false, a, false, b1, (short)0, c1, false, false);
        c2 = __builtin_amdgcn_wmma_f32_16x16x32_f16(false, a, false, b2, (short)0, c2, false, false);
        c3 = __builtin_amdgcn_wmma_f32_16x16x32_f16(false, a, false, b3, (short)0, c3, false, false);
    }

    // ---- store D + bias: VGPR r -> row m0 + r + (lane>=16 ? 8 : 0), col = lane&15
    int col  = lane & 15;
    int mrow = m0 + ((lane & 16) ? 8 : 0);
    float bv0 = bias[col], bv1 = bias[16 + col], bv2 = bias[32 + col], bv3 = bias[48 + col];
#pragma unroll
    for (int r = 0; r < 8; ++r) {
        int row = mrow + r;
        if (row < M) {
            float* po = out + (size_t)row * DOUT;
            po[col]      = c0[r] + bv0;
            po[16 + col] = c1[r] + bv1;
            po[32 + col] = c2[r] + bv2;
            po[48 + col] = c3[r] + bv3;
        }
    }
}

// ---------------------------------------------------------------------------
// init per-layer accumulators: smax = -inf, denom = 0, agg = 0
// ---------------------------------------------------------------------------
__global__ void init_accum(float* __restrict__ smax, float* __restrict__ denom,
                           float* __restrict__ agg, int Nn) {
    int i = (int)(blockIdx.x * blockDim.x + threadIdx.x);
    int total = Nn * DOUT;
    if (i < total) agg[i] = 0.0f;
    if (i < Nn) {
        smax[i]  = -INFINITY;
        denom[i] = 0.0f;
    }
}

// ---------------------------------------------------------------------------
// Pass 1: per-edge attention logit, one wave per edge (2 elems / lane).
// s[e] = leaky_relu(xl[dst] + xr[src]) . att   ; smax[src] = max(...)
// Edges e >= E are self-loops (src = dst = e - E).
// ---------------------------------------------------------------------------
__global__ void edge_logits(const int* __restrict__ esrc, const int* __restrict__ edst,
                            const float* __restrict__ xl, const float* __restrict__ xr,
                            const float* __restrict__ att,
                            float* __restrict__ s, float* __restrict__ smax,
                            int E, int ET) {
    int wave = (int)((blockIdx.x * blockDim.x + threadIdx.x) >> 5);
    int lane = (int)(threadIdx.x & 31);
    if (wave >= ET) return;
    int src, dst;
    if (wave < E) { src = esrc[wave]; dst = edst[wave]; }
    else          { src = dst = wave - E; }

    const float* pl = xl + (size_t)dst * DOUT;
    const float* pr = xr + (size_t)src * DOUT;
    float p = 0.0f;
#pragma unroll
    for (int j = 0; j < 2; ++j) {
        int idx = lane + j * 32;
        float m = pl[idx] + pr[idx];
        m = (m > 0.0f) ? m : 0.2f * m;        // leaky_relu, NEG_SLOPE=0.2
        p += m * att[idx];
    }
#pragma unroll
    for (int off = 16; off > 0; off >>= 1) p += __shfl_xor(p, off, 32);
    if (lane == 0) {
        s[wave] = p;
        atomicMaxFloat(&smax[src], p);
    }
}

// ---------------------------------------------------------------------------
// Pass 2: s[e] = exp(s[e] - smax[src]);  denom[src] += s[e]   (1 thread/edge)
// ---------------------------------------------------------------------------
__global__ void edge_exp(const int* __restrict__ esrc,
                         float* __restrict__ s, const float* __restrict__ smax,
                         float* __restrict__ denom, int E, int ET) {
    int e = (int)(blockIdx.x * blockDim.x + threadIdx.x);
    if (e >= ET) return;
    int src = (e < E) ? esrc[e] : (e - E);
    float v = __expf(s[e] - smax[src]);
    s[e] = v;
    atomicAdd(&denom[src], v);
}

// ---------------------------------------------------------------------------
// Pass 3: agg[src] += (s[e]/denom[src]) * xl[dst]   (1 wave/edge, coalesced)
// ---------------------------------------------------------------------------
__global__ void edge_aggregate(const int* __restrict__ esrc, const int* __restrict__ edst,
                               const float* __restrict__ xl, const float* __restrict__ s,
                               const float* __restrict__ denom,
                               float* __restrict__ agg, int E, int ET) {
    int wave = (int)((blockIdx.x * blockDim.x + threadIdx.x) >> 5);
    int lane = (int)(threadIdx.x & 31);
    if (wave >= ET) return;
    int src, dst;
    if (wave < E) { src = esrc[wave]; dst = edst[wave]; }
    else          { src = dst = wave - E; }

    float alpha = s[wave] / denom[src];
    const float* pl = xl + (size_t)dst * DOUT;
    float*       po = agg + (size_t)src * DOUT;
    atomicAdd(&po[lane],      alpha * pl[lane]);
    atomicAdd(&po[lane + 32], alpha * pl[lane + 32]);
}

// ---------------------------------------------------------------------------
// Final per-layer epilogue: out = relu(agg + bias)
// ---------------------------------------------------------------------------
__global__ void bias_relu(const float* __restrict__ agg, const float* __restrict__ bias,
                          float* __restrict__ out, int Nn) {
    int i = (int)(blockIdx.x * blockDim.x + threadIdx.x);
    if (i >= Nn * DOUT) return;
    float v = agg[i] + bias[i & (DOUT - 1)];
    out[i] = (v > 0.0f) ? v : 0.0f;
}

// ---------------------------------------------------------------------------
// host launcher
// ---------------------------------------------------------------------------
extern "C" void kernel_launch(void* const* d_in, const int* in_sizes, int n_in,
                              void* d_out, int out_size, void* d_ws, size_t ws_size,
                              hipStream_t stream) {
    const float* x  = (const float*)d_in[0];
    const int*   ei = (const int*)d_in[1];
    // d_in[2] = edge_attr (unused by the module)

    const int F_IN = 32;
    int Nn = in_sizes[0] / F_IN;       // 100000
    int E  = in_sizes[1] / 2;          // 1600000
    int ET = E + Nn;                   // + self loops
    const int* esrc = ei;
    const int* edst = ei + E;

    const float* Wl[3]  = {(const float*)d_in[3],  (const float*)d_in[9],  (const float*)d_in[15]};
    const float* bl[3]  = {(const float*)d_in[4],  (const float*)d_in[10], (const float*)d_in[16]};
    const float* Wr[3]  = {(const float*)d_in[5],  (const float*)d_in[11], (const float*)d_in[17]};
    const float* br[3]  = {(const float*)d_in[6],  (const float*)d_in[12], (const float*)d_in[18]};
    const float* att[3] = {(const float*)d_in[7],  (const float*)d_in[13], (const float*)d_in[19]};
    const float* bb[3]  = {(const float*)d_in[8],  (const float*)d_in[14], (const float*)d_in[20]};

    // workspace layout (f32)
    float* ws    = (float*)d_ws;
    size_t nd    = (size_t)Nn * DOUT;
    float* xl    = ws;                 // [Nn*64]
    float* xr    = xl + nd;            // [Nn*64]
    float* agg   = xr + nd;            // [Nn*64]
    float* hbuf  = agg + nd;           // [Nn*64]
    float* s     = hbuf + nd;          // [ET]
    float* smax  = s + ET;             // [Nn]
    float* denom = smax + Nn;          // [Nn]

    int gemm_waves  = (Nn + 15) / 16;
    int gemm_blocks = (gemm_waves * 32 + TPB - 1) / TPB;
    int node_blocks = (Nn * DOUT + TPB - 1) / TPB;
    int edgew_blocks = (int)(((long long)ET * 32 + TPB - 1) / TPB);
    int edget_blocks = (ET + TPB - 1) / TPB;

    for (int li = 0; li < 3; ++li) {
        const float* hin = (li == 0) ? x : hbuf;
        int K = (li == 0) ? F_IN : DOUT;

        gemm_bias_wmma<<<gemm_blocks, TPB, 0, stream>>>(hin, Wl[li], bl[li], xl, Nn, K);
        gemm_bias_wmma<<<gemm_blocks, TPB, 0, stream>>>(hin, Wr[li], br[li], xr, Nn, K);
        init_accum<<<node_blocks, TPB, 0, stream>>>(smax, denom, agg, Nn);
        edge_logits<<<edgew_blocks, TPB, 0, stream>>>(esrc, edst, xl, xr, att[li], s, smax, E, ET);
        edge_exp<<<edget_blocks, TPB, 0, stream>>>(esrc, s, smax, denom, E, ET);
        edge_aggregate<<<edgew_blocks, TPB, 0, stream>>>(esrc, edst, xl, s, denom, agg, E, ET);

        float* hout = (li == 2) ? (float*)d_out : hbuf;
        bias_relu<<<node_blocks, TPB, 0, stream>>>(agg, bb[li], hout, Nn);
    }
}
